// AttentionLayer_32667521254136
// MI455X (gfx1250) — matrix operand order repaired and checked by reading, not compiled, not run on hardware
//
#include <hip/hip_runtime.h>

// Attention layer: B=4, S=2048, D=1024, H=16, HD=64.
// bf16 WMMA (v_wmma_f32_16x16x32_bf16), f32 accumulation, flash-style
// online softmax, async global->LDS staging of K/V tiles (ASYNCcnt path).

#define BB   4
#define SS   2048
#define DDIM 1024
#define HH   16
#define HDIM 64

typedef __attribute__((ext_vector_type(16))) __bf16 v16bf;
typedef __attribute__((ext_vector_type(8)))  __bf16 bf16x8;
typedef __attribute__((ext_vector_type(8)))  float  v8f;
typedef unsigned int u32;
typedef unsigned short u16;

// ---- async global->LDS copy (CDNA5 GLOBAL_LOAD_ASYNC_TO_LDS_B128) ----------
#if defined(__has_builtin)
#if __has_builtin(__builtin_amdgcn_global_load_async_to_lds_b128)
#define HAS_ASYNC_LDS 1
#endif
#endif

#ifdef HAS_ASYNC_LDS
typedef int v4i_ __attribute__((vector_size(16)));
typedef __attribute__((address_space(1))) v4i_ gas_v4i;   // global
typedef __attribute__((address_space(3))) v4i_ las_v4i;   // LDS
static __device__ __forceinline__ void copy_b128(void* lds_dst, const void* gsrc) {
  // flat-address low 32 bits == LDS offset (ISA 10.2)
  __builtin_amdgcn_global_load_async_to_lds_b128(
      (gas_v4i*)(unsigned long long)(size_t)gsrc,
      (las_v4i*)(unsigned int)(size_t)lds_dst,
      0, 0);
}
#define WAIT_ASYNC_4() asm volatile("s_wait_asynccnt 0x4" ::: "memory")
#define WAIT_ASYNC_0() asm volatile("s_wait_asynccnt 0x0" ::: "memory")
#else
static __device__ __forceinline__ void copy_b128(void* lds_dst, const void* gsrc) {
  *(uint4*)lds_dst = *(const uint4*)gsrc;   // sync fallback
}
#define WAIT_ASYNC_4()
#define WAIT_ASYNC_0()
#endif

static __device__ __forceinline__ u16 f2bf_bits(float f) {
  u32 u = __builtin_bit_cast(u32, f);
  u32 r = u + 0x7fffu + ((u >> 16) & 1u);   // round-to-nearest-even
  return (u16)(r >> 16);
}

static __device__ __forceinline__ v8f wmma_bf16(v16bf a, v16bf b, v8f c) {
  return __builtin_amdgcn_wmma_f32_16x16x32_bf16(
      false, a, false, b, (short)0, c, false, false);
}

static __device__ __forceinline__ v16bf ld_v16(const u16* p) {
  return *(const v16bf*)p;
}

// ---------------------------------------------------------------------------
// Stage 1: fp32 -> bf16 conversion, 8 elements per thread
// ---------------------------------------------------------------------------
__global__ void cvt_kernel(const float* __restrict__ s, u16* __restrict__ d, int n8) {
  int i = blockIdx.x * blockDim.x + threadIdx.x;
  if (i >= n8) return;
  float4 a = ((const float4*)s)[2 * i];
  float4 b = ((const float4*)s)[2 * i + 1];
  u32 o0 = (u32)f2bf_bits(a.x) | ((u32)f2bf_bits(a.y) << 16);
  u32 o1 = (u32)f2bf_bits(a.z) | ((u32)f2bf_bits(a.w) << 16);
  u32 o2 = (u32)f2bf_bits(b.x) | ((u32)f2bf_bits(b.y) << 16);
  u32 o3 = (u32)f2bf_bits(b.z) | ((u32)f2bf_bits(b.w) << 16);
  ((uint4*)d)[i] = make_uint4(o0, o1, o2, o3);
}

// ---------------------------------------------------------------------------
// Stage 2: QKV projection, y = x @ W^T + b.
// One wave computes a 32(M) x 64(N) tile: each B (weight) fragment feeds two
// WMMAs -> 8 wmma per K=32 chunk from 12 b128 loads.
// Q,K stored bf16 [B*S, D]; V stored transposed bf16 [B,H,HD,S].
// ---------------------------------------------------------------------------
__global__ void __launch_bounds__(128) qkv_kernel(
    const __bf16* __restrict__ xb,
    const __bf16* __restrict__ wq, const __bf16* __restrict__ wk, const __bf16* __restrict__ wv,
    const float* __restrict__ bq, const float* __restrict__ bk, const float* __restrict__ bv,
    u16* __restrict__ qb, u16* __restrict__ kb, u16* __restrict__ vt)
{
  const int lane  = threadIdx.x & 31;
  const int warp  = threadIdx.x >> 5;
  const int w     = blockIdx.x * 4 + warp;
  const int mt    = w / 48;           // 256 M-tiles of 32 rows over B*S=8192
  const int rem   = w % 48;
  const int mat   = rem / 16;         // 0=Q, 1=K, 2=V
  const int nt    = rem % 16;         // 64-wide N tile
  const int lmod  = lane & 15;
  const int lhalf = lane >> 4;
  const int m0    = mt * 32;

  const __bf16* W    = (mat == 0) ? wq : (mat == 1) ? wk : wv;
  const float*  bias = (mat == 0) ? bq : (mat == 1) ? bk : bv;

  const __bf16* arow0 = xb + (size_t)(m0 + lmod) * DDIM + 16 * lhalf;
  const __bf16* arow1 = arow0 + (size_t)16 * DDIM;
  const __bf16* bcol[4];
#pragma unroll
  for (int j = 0; j < 4; ++j)
    bcol[j] = W + (size_t)(nt * 64 + j * 16 + lmod) * DDIM + 16 * lhalf;

  v8f acc[2][4];
#pragma unroll
  for (int i = 0; i < 2; ++i)
#pragma unroll
    for (int j = 0; j < 4; ++j) acc[i][j] = v8f{};

  for (int kc = 0; kc < DDIM; kc += 32) {
    bf16x8 lo0 = *(const bf16x8*)(arow0 + kc);
    bf16x8 hi0 = *(const bf16x8*)(arow0 + kc + 8);
    bf16x8 lo1 = *(const bf16x8*)(arow1 + kc);
    bf16x8 hi1 = *(const bf16x8*)(arow1 + kc + 8);
    v16bf a0, a1;
#pragma unroll
    for (int i = 0; i < 8; ++i) {
      a0[i] = lo0[i]; a0[i + 8] = hi0[i];
      a1[i] = lo1[i]; a1[i + 8] = hi1[i];
    }
#pragma unroll
    for (int j = 0; j < 4; ++j) {
      v16bf f = *(const v16bf*)(bcol[j] + kc);
      acc[0][j] = wmma_bf16(a0, f, acc[0][j]);
      acc[1][j] = wmma_bf16(a1, f, acc[1][j]);
    }
  }

#pragma unroll
  for (int j = 0; j < 4; ++j) {
    const int   n  = nt * 64 + j * 16 + lmod;
    const float bn = bias[n];
#pragma unroll
    for (int i = 0; i < 2; ++i) {
      if (mat < 2) {
        u16* outp = (mat == 0) ? qb : kb;
#pragma unroll
        for (int r = 0; r < 8; ++r) {
          int row = m0 + i * 16 + r + 8 * lhalf;
          outp[(size_t)row * DDIM + n] = f2bf_bits(acc[i][j][r] + bn);
        }
      } else {
        const int h  = nt;
        const int hd = j * 16 + lmod;
#pragma unroll
        for (int r = 0; r < 8; ++r) {
          int row = m0 + i * 16 + r + 8 * lhalf;
          int bi  = row >> 11;
          int si  = row & (SS - 1);
          vt[(((size_t)bi * HH + h) * HDIM + hd) * SS + si] = f2bf_bits(acc[i][j][r] + bn);
        }
      }
    }
  }
}

// ---------------------------------------------------------------------------
// Stage 3: flash attention. One block = 4 waves = 4 consecutive 16-query
// tiles of the SAME (b,h): K/V tiles are shared, so they are staged once per
// block into LDS with async global->LDS copies, ping-pong double buffered.
// ---------------------------------------------------------------------------
__global__ void __launch_bounds__(128) attn_kernel(
    const __bf16* __restrict__ qb, const __bf16* __restrict__ kb,
    const __bf16* __restrict__ vt, float* __restrict__ out)
{
  // K tile: 32 keys x 64 hd, row stride 72 halves (144B) -> conflict-free b128 reads
  // V tile: 64 hd  x 32 k,  row stride 40 halves (80B)  -> conflict-free b128 reads
  // P tile: per-wave 16x32, row stride 40 halves
  __shared__ __align__(16) u16 sK[2][32 * 72];
  __shared__ __align__(16) u16 sV[2][64 * 40];
  __shared__ __align__(16) u16 sp_all[4][16 * 40];

  const int tid   = threadIdx.x;
  const int lane  = tid & 31;
  const int warp  = tid >> 5;
  u16* sp = sp_all[warp];

  const int w     = blockIdx.x * 4 + warp;
  const int qt    = w & 127;           // 128 query tiles per (b,h); same (b,h) per block
  const int bh    = w >> 7;
  const int b     = bh >> 4;
  const int h     = bh & 15;
  const int lmod  = lane & 15;
  const int lhalf = lane >> 4;

  // ---- Q A-fragments kept in registers (HD=64 -> two K=32 chunks)
  v16bf aq0, aq1;
  {
    const __bf16* qrow = qb + (size_t)(b * SS + qt * 16 + lmod) * DDIM + h * HDIM + 16 * lhalf;
    bf16x8 lo = *(const bf16x8*)(qrow);
    bf16x8 hi = *(const bf16x8*)(qrow + 8);
#pragma unroll
    for (int i = 0; i < 8; ++i) { aq0[i] = lo[i]; aq0[i + 8] = hi[i]; }
    lo = *(const bf16x8*)(qrow + 32);
    hi = *(const bf16x8*)(qrow + 40);
#pragma unroll
    for (int i = 0; i < 8; ++i) { aq1[i] = lo[i]; aq1[i + 8] = hi[i]; }
  }

  float mrow[8], lrow[8];
#pragma unroll
  for (int r = 0; r < 8; ++r) { mrow[r] = -__builtin_inff(); lrow[r] = 0.0f; }
  v8f o[4] = {v8f{}, v8f{}, v8f{}, v8f{}};

  // ---- cooperative tile staging: 256 K-copies + 256 V-copies of 16B, 4/thread
  const __bf16* kbase = kb + (size_t)(b * SS) * DDIM + h * HDIM;
  const __bf16* vbase = vt + ((size_t)b * HH + h) * (size_t)HDIM * SS;
  const int kcA = tid, kcB = tid + 128;          // K copy ids: row=c>>3, seg=c&7
  const int vcA = tid, vcB = tid + 128;          // V copy ids: row=c>>2, seg=c&3

  auto issue_tile = [&](int buf, int kt) {
    copy_b128(&sK[buf][(kcA >> 3) * 72 + (kcA & 7) * 8],
              kbase + (size_t)(kt + (kcA >> 3)) * DDIM + (kcA & 7) * 8);
    copy_b128(&sK[buf][(kcB >> 3) * 72 + (kcB & 7) * 8],
              kbase + (size_t)(kt + (kcB >> 3)) * DDIM + (kcB & 7) * 8);
    copy_b128(&sV[buf][(vcA >> 2) * 40 + (vcA & 3) * 8],
              vbase + (size_t)(vcA >> 2) * SS + kt + (vcA & 3) * 8);
    copy_b128(&sV[buf][(vcB >> 2) * 40 + (vcB & 3) * 8],
              vbase + (size_t)(vcB >> 2) * SS + kt + (vcB & 3) * 8);
  };

  const float scale = 0.125f;          // 1/sqrt(64)

  issue_tile(0, 0);
  for (int kt = 0, it = 0; kt < SS; kt += 32, ++it) {
    const int cur = it & 1;
    if (kt + 32 < SS) {
      issue_tile(cur ^ 1, kt + 32);
      WAIT_ASYNC_4();                  // 4 in-flight ops belong to next tile
    } else {
      WAIT_ASYNC_0();
    }
    __syncthreads();                   // current tile resident for all waves

    const u16* Kt = sK[cur];
    const u16* Vt = sV[cur];

    // ---- scores: two 16x16 tiles over keys [kt, kt+32)
    v8f s0 = {}, s1 = {};
    s0 = wmma_bf16(aq0, ld_v16(&Kt[(lmod)      * 72 +      16 * lhalf]), s0);
    s0 = wmma_bf16(aq1, ld_v16(&Kt[(lmod)      * 72 + 32 + 16 * lhalf]), s0);
    s1 = wmma_bf16(aq0, ld_v16(&Kt[(16 + lmod) * 72 +      16 * lhalf]), s1);
    s1 = wmma_bf16(aq1, ld_v16(&Kt[(16 + lmod) * 72 + 32 + 16 * lhalf]), s1);

    // ---- online softmax update (row = vgpr index r + 8*lhalf)
    float p0[8], p1[8], alpha[8];
#pragma unroll
    for (int r = 0; r < 8; ++r) {
      float a_ = s0[r] * scale;
      float b_ = s1[r] * scale;
      float mx = fmaxf(a_, b_);
      mx = fmaxf(mx, __shfl_xor(mx, 1, 32));
      mx = fmaxf(mx, __shfl_xor(mx, 2, 32));
      mx = fmaxf(mx, __shfl_xor(mx, 4, 32));
      mx = fmaxf(mx, __shfl_xor(mx, 8, 32));
      float mnew = fmaxf(mrow[r], mx);
      alpha[r] = __expf(mrow[r] - mnew);
      mrow[r]  = mnew;
      p0[r] = __expf(a_ - mnew);
      p1[r] = __expf(b_ - mnew);
      float rs = p0[r] + p1[r];
      rs += __shfl_xor(rs, 1, 32);
      rs += __shfl_xor(rs, 2, 32);
      rs += __shfl_xor(rs, 4, 32);
      rs += __shfl_xor(rs, 8, 32);
      lrow[r] = lrow[r] * alpha[r] + rs;
    }
#pragma unroll
    for (int r = 0; r < 8; ++r) {
      o[0][r] *= alpha[r]; o[1][r] *= alpha[r];
      o[2][r] *= alpha[r]; o[3][r] *= alpha[r];
    }

    // ---- P (C layout) -> LDS bf16, then reload as A-operand (same-wave swizzle)
#pragma unroll
    for (int r = 0; r < 8; ++r) {
      int row = r + 8 * lhalf;
      sp[row * 40 + lmod]      = f2bf_bits(p0[r]);
      sp[row * 40 + 16 + lmod] = f2bf_bits(p1[r]);
    }
    v16bf ap;
    {
      const u16* base = sp + lmod * 40 + 16 * lhalf;
      bf16x8 lo = *(const bf16x8*)(base);
      bf16x8 hi = *(const bf16x8*)(base + 8);
#pragma unroll
      for (int i = 0; i < 8; ++i) { ap[i] = lo[i]; ap[i + 8] = hi[i]; }
    }

    // ---- P @ V from the shared V tile
#pragma unroll
    for (int t = 0; t < 4; ++t) {
      v16bf bv = ld_v16(&Vt[(t * 16 + lmod) * 40 + 16 * lhalf]);
      o[t] = wmma_bf16(ap, bv, o[t]);
    }

    __syncthreads();                   // done reading: buffer may be overwritten
  }

  // ---- epilogue: O / l, fp32 store to [B,S,D]
#pragma unroll
  for (int r = 0; r < 8; ++r) {
    float invl = 1.0f / lrow[r];
    int row = qt * 16 + r + 8 * lhalf;
    float* orow = out + (size_t)(b * SS + row) * DDIM + h * HDIM + lmod;
    orow[0]  = o[0][r] * invl;
    orow[16] = o[1][r] * invl;
    orow[32] = o[2][r] * invl;
    orow[48] = o[3][r] * invl;
  }
}

// ---------------------------------------------------------------------------
extern "C" void kernel_launch(void* const* d_in, const int* in_sizes, int n_in,
                              void* d_out, int out_size, void* d_ws, size_t ws_size,
                              hipStream_t stream)
{
  const float* x  = (const float*)d_in[0];
  const float* Wq = (const float*)d_in[1];
  const float* bq = (const float*)d_in[2];
  const float* Wk = (const float*)d_in[3];
  const float* bk = (const float*)d_in[4];
  const float* Wv = (const float*)d_in[5];
  const float* bv = (const float*)d_in[6];
  float* out = (float*)d_out;

  char* ws = (char*)d_ws;
  const size_t SZ_X = (size_t)BB * SS * DDIM * 2;   // 16 MB bf16
  const size_t SZ_W = (size_t)DDIM * DDIM * 2;      //  2 MB bf16
  __bf16* xb  = (__bf16*)(ws);
  __bf16* wqb = (__bf16*)(ws + SZ_X);
  __bf16* wkb = (__bf16*)(ws + SZ_X + SZ_W);
  __bf16* wvb = (__bf16*)(ws + SZ_X + 2 * SZ_W);
  __bf16* qb  = (__bf16*)(ws + SZ_X + 3 * SZ_W);
  __bf16* kb  = (__bf16*)(ws + 2 * SZ_X + 3 * SZ_W);
  __bf16* vt  = (__bf16*)(ws + 3 * SZ_X + 3 * SZ_W);

  const int nx8 = BB * SS * DDIM / 8;
  const int nw8 = DDIM * DDIM / 8;
  cvt_kernel<<<(nx8 + 255) / 256, 256, 0, stream>>>(x,  (u16*)xb,  nx8);
  cvt_kernel<<<(nw8 + 255) / 256, 256, 0, stream>>>(Wq, (u16*)wqb, nw8);
  cvt_kernel<<<(nw8 + 255) / 256, 256, 0, stream>>>(Wk, (u16*)wkb, nw8);
  cvt_kernel<<<(nw8 + 255) / 256, 256, 0, stream>>>(Wv, (u16*)wvb, nw8);

  // 256 M-tiles * 16 N-tiles * 3 matrices = 12288 waves / 4 per block
  qkv_kernel<<<3072, 128, 0, stream>>>(xb, wqb, wkb, wvb, bq, bk, bv,
                                       (u16*)qb, (u16*)kb, (u16*)vt);

  // B*H*(S/16) = 8192 waves / 4 per block
  attn_kernel<<<2048, 128, 0, stream>>>(qb, kb, vt, out);
}
